// SPoTrPartDecoder_38422777430269
// MI455X (gfx1250) — matrix operand order repaired and checked by reading, not compiled, not run on hardware
//
#include <hip/hip_runtime.h>
#include <hip/hip_bf16.h>

// ---------------------------------------------------------------------------
// SPoTrPartDecoder for MI455X (gfx1250)
//   - bf16 WMMA GEMM core (v_wmma_f32_16x16x32_bf16, f32 accumulate)
//   - activations kept bf16 in [B][N][K] (K innermost, K padded to mult of 64)
//   - weights pre-converted to bf16 [M][Kpad]
//   - GEMM is branch-free: uniform b128 global->LDS staging, no guards
// ---------------------------------------------------------------------------

#define BATCH 4
#define NCLS  50

typedef __attribute__((ext_vector_type(16))) __bf16 v16bf;
typedef __attribute__((ext_vector_type(8)))  __bf16 v8bf;
typedef __attribute__((ext_vector_type(8)))  float  v8f;

__device__ inline v16bf cat8(v8bf lo, v8bf hi) {
  return __builtin_shufflevector(lo, hi, 0,1,2,3,4,5,6,7,8,9,10,11,12,13,14,15);
}

// ---------------------------------------------------------------------------
// Global class embeddings: emb[b][o] = max_n relu(W.f + b)
// ---------------------------------------------------------------------------
__global__ __launch_bounds__(256)
void embed_kernel(const float* __restrict__ f3, const float* __restrict__ f4,
                  const float* __restrict__ w1, const float* __restrict__ b1,
                  const float* __restrict__ w2, const float* __restrict__ b2,
                  float* __restrict__ emb1, float* __restrict__ emb2) {
  int b = blockIdx.x;
  int t = threadIdx.x;
  if (t < 64) {                       // emb1: f3 [B,512,128] -> [B,64]
    float best = 0.0f;
    const float* w = w1 + (size_t)t * 512;
    for (int n = 0; n < 128; ++n) {
      float acc = b1[t];
      const float* f = f3 + (size_t)b * 512 * 128 + n;
      for (int i = 0; i < 512; ++i) acc += w[i] * f[(size_t)i * 128];
      best = fmaxf(best, fmaxf(acc, 0.0f));
    }
    emb1[b * 64 + t] = best;
  } else if (t < 192) {               // emb2: f4 [B,1024,32] -> [B,128]
    int o = t - 64;
    float best = 0.0f;
    const float* w = w2 + (size_t)o * 1024;
    for (int n = 0; n < 32; ++n) {
      float acc = b2[o];
      const float* f = f4 + (size_t)b * 1024 * 32 + n;
      for (int i = 0; i < 1024; ++i) acc += w[i] * f[(size_t)i * 32];
      best = fmaxf(best, fmaxf(acc, 0.0f));
    }
    emb2[b * 128 + o] = best;
  }
}

// ---------------------------------------------------------------------------
// 3-NN search + inverse-distance weights (matches lax.top_k tie order)
// ---------------------------------------------------------------------------
__global__ __launch_bounds__(256)
void knn3_kernel(const float* __restrict__ p1, const float* __restrict__ p2,
                 int N1, int N2, int* __restrict__ idx, float* __restrict__ wgt) {
  int t = blockIdx.x * blockDim.x + threadIdx.x;
  if (t >= BATCH * N1) return;
  int b = t / N1, n = t - b * N1;
  const float* q = p1 + ((size_t)b * N1 + n) * 3;
  float qx = q[0], qy = q[1], qz = q[2];
  float d0 = 3.4e38f, d1 = 3.4e38f, d2 = 3.4e38f;
  int i0 = 0, i1 = 0, i2 = 0;
  const float* src = p2 + (size_t)b * N2 * 3;
  for (int m = 0; m < N2; ++m) {
    float dx = qx - src[m * 3 + 0];
    float dy = qy - src[m * 3 + 1];
    float dz = qz - src[m * 3 + 2];
    float d = dx * dx + dy * dy + dz * dz;
    if (d < d0)      { d2 = d1; i2 = i1; d1 = d0; i1 = i0; d0 = d; i0 = m; }
    else if (d < d1) { d2 = d1; i2 = i1; d1 = d;  i1 = m; }
    else if (d < d2) { d2 = d;  i2 = m; }
  }
  float w0 = 1.0f / (fmaxf(d0, 0.0f) + 1e-8f);
  float w1 = 1.0f / (fmaxf(d1, 0.0f) + 1e-8f);
  float w2 = 1.0f / (fmaxf(d2, 0.0f) + 1e-8f);
  float s = 1.0f / (w0 + w1 + w2);
  size_t base = (size_t)t * 3;
  idx[base + 0] = i0; idx[base + 1] = i1; idx[base + 2] = i2;
  wgt[base + 0] = w0 * s; wgt[base + 1] = w1 * s; wgt[base + 2] = w2 * s;
}

// ---------------------------------------------------------------------------
// Weight conversion: f32 [M][K] -> bf16 [M][Kp] (zero-padded)
// ---------------------------------------------------------------------------
__global__ __launch_bounds__(256)
void convert_w_kernel(const float* __restrict__ W, __bf16* __restrict__ Wb,
                      int M, int K, int Kp) {
  int t = blockIdx.x * blockDim.x + threadIdx.x;
  if (t >= M * Kp) return;
  int m = t / Kp, k = t - m * Kp;
  Wb[t] = (k < K) ? (__bf16)W[(size_t)m * K + k] : (__bf16)0.0f;
}

// ---------------------------------------------------------------------------
// Build X = concat(skip, interp(fsrc)) as bf16 [B][N1][Kp]  (K innermost)
//   skip:  f32 [B][C1][N1]
//   fsrc:  srcNC ? bf16 [B][N2][Cs] (prev stage out) : f32 [B][C2][N2]
// ---------------------------------------------------------------------------
__global__ __launch_bounds__(256)
void build_x_kernel(const float* __restrict__ skip, int C1,
                    const void* __restrict__ fsrc, int C2, int srcNC, int Cs,
                    const int* __restrict__ idx, const float* __restrict__ wgt,
                    int N1, int N2, int Kp, __bf16* __restrict__ X) {
  size_t total = (size_t)BATCH * N1 * Kp;
  size_t t = (size_t)blockIdx.x * blockDim.x + threadIdx.x;
  if (t >= total) return;
  int k = (int)(t % Kp);
  size_t bn = t / Kp;
  int n = (int)(bn % N1);
  int b = (int)(bn / N1);
  float v = 0.0f;
  if (k < C1) {
    v = skip[((size_t)b * C1 + k) * N1 + n];
  } else if (k < C1 + C2) {
    int c = k - C1;
    size_t base = ((size_t)b * N1 + n) * 3;
    if (srcNC) {
      const __bf16* f = (const __bf16*)fsrc + (size_t)b * N2 * Cs + c;
      v = wgt[base + 0] * (float)f[(size_t)idx[base + 0] * Cs]
        + wgt[base + 1] * (float)f[(size_t)idx[base + 1] * Cs]
        + wgt[base + 2] * (float)f[(size_t)idx[base + 2] * Cs];
    } else {
      const float* f = (const float*)fsrc + ((size_t)b * C2 + c) * N2;
      v = wgt[base + 0] * f[idx[base + 0]]
        + wgt[base + 1] * f[idx[base + 1]]
        + wgt[base + 2] * f[idx[base + 2]];
    }
  }
  X[t] = (__bf16)v;
}

// Stage-0 concat: [emb1(64)|emb2(128)|onehot(50)|f0(64)|interp(f1u,128)] pad->448
__global__ __launch_bounds__(256)
void build_x0_kernel(const float* __restrict__ emb1, const float* __restrict__ emb2,
                     const int* __restrict__ cls, const float* __restrict__ f0,
                     const __bf16* __restrict__ fsrc, /*Cs=*/int Cs,
                     const int* __restrict__ idx, const float* __restrict__ wgt,
                     int N1, int N2, int Kp, __bf16* __restrict__ X) {
  size_t total = (size_t)BATCH * N1 * Kp;
  size_t t = (size_t)blockIdx.x * blockDim.x + threadIdx.x;
  if (t >= total) return;
  int k = (int)(t % Kp);
  size_t bn = t / Kp;
  int n = (int)(bn % N1);
  int b = (int)(bn / N1);
  float v = 0.0f;
  if (k < 64) {
    v = emb1[b * 64 + k];
  } else if (k < 192) {
    v = emb2[b * 128 + (k - 64)];
  } else if (k < 242) {
    v = (cls[b] == (k - 192)) ? 1.0f : 0.0f;
  } else if (k < 306) {
    v = f0[((size_t)b * 64 + (k - 242)) * N1 + n];
  } else if (k < 434) {
    int c = k - 306;
    size_t base = ((size_t)b * N1 + n) * 3;
    const __bf16* f = fsrc + (size_t)b * N2 * Cs + c;
    v = wgt[base + 0] * (float)f[(size_t)idx[base + 0] * Cs]
      + wgt[base + 1] * (float)f[(size_t)idx[base + 1] * Cs]
      + wgt[base + 2] * (float)f[(size_t)idx[base + 2] * Cs];
  }
  X[t] = (__bf16)v;
}

// ---------------------------------------------------------------------------
// GEMM + bias + BN + ReLU via v_wmma_f32_16x16x32_bf16. Branch-free.
//   X  : bf16 [B][N][K]   (K = padded, multiple of 64)
//   Wb : bf16 [M][K]
//   Y  : OUT_F32_MN ? f32 [B][M][N] : bf16 [B][N][M]
// Block 256 = 8 wave32, 64x64 tile; wave = 16x32 sub-tile; BK=64 ->
// 4 WMMAs per barrier pair; all LDS traffic is b128.
// ---------------------------------------------------------------------------
#define BM 64
#define BN 64
#define BK 64
#define LDP 72   // 144B row stride: 16B-aligned, bank-spread

template <bool OUT_F32_MN>
__global__ __launch_bounds__(256)
void gemm_bn_relu_kernel(const __bf16* __restrict__ X,
                         const __bf16* __restrict__ Wb,
                         const float* __restrict__ bias,
                         const float* __restrict__ g,
                         const float* __restrict__ be,
                         void* __restrict__ Yv,
                         int M, int K, int N) {
  __shared__ __align__(16) __bf16 sA[BM][LDP];  // weights [m][k]
  __shared__ __align__(16) __bf16 sB[BN][LDP];  // acts    [n][k]

  int b = blockIdx.z;
  int n_block = blockIdx.x * BN;
  int m_block = blockIdx.y * BM;
  const __bf16* Xb = X + (size_t)b * N * K;

  int tid  = threadIdx.x;
  int wave = tid >> 5, lane = tid & 31;
  int wm = wave >> 1, wn = wave & 1;
  int half = lane >> 4, l15 = lane & 15;

  v8f acc0 = {}; v8f acc1 = {};

  for (int k0 = 0; k0 < K; k0 += BK) {
    // uniform staging: 1024 b128 chunks (A:512, B:512), 2 chunks/thread each
#pragma unroll
    for (int c = 0; c < 2; ++c) {
      int id = tid + c * 256;          // 0..511
      int r  = id >> 3;                // row 0..63
      int ko = (id & 7) * 8;           // 0..56
      *(v8bf*)&sA[r][ko] = *(const v8bf*)&Wb[(size_t)(m_block + r) * K + k0 + ko];
      *(v8bf*)&sB[r][ko] = *(const v8bf*)&Xb[(size_t)(n_block + r) * K + k0 + ko];
    }
    if (k0 + BK < K) {                 // prefetch next K-tile (global_prefetch_b8)
      int r  = tid >> 3;
      int ko = (tid & 7) * 8;
      __builtin_prefetch(&Wb[(size_t)(m_block + r) * K + k0 + BK + ko], 0, 1);
      __builtin_prefetch(&Xb[(size_t)(n_block + r) * K + k0 + BK + ko], 0, 1);
    }
    __syncthreads();

#pragma unroll
    for (int s = 0; s < 2; ++s) {      // two 16x16x32 K-substeps
      int ks  = s * 32;
      int ka0 = ks + (half ? 8 : 0);
      int ka1 = ks + (half ? 24 : 16);
      const __bf16* arow = &sA[wm * 16 + l15][0];
      v16bf af = cat8(*(const v8bf*)(arow + ka0), *(const v8bf*)(arow + ka1));

      int kb = ks + (half ? 8 : 0);
      const __bf16* brow0 = &sB[wn * 32 + l15][kb];
      const __bf16* brow1 = &sB[wn * 32 + 16 + l15][kb];
      v16bf bf0 = cat8(*(const v8bf*)brow0, *(const v8bf*)(brow0 + 16));
      v16bf bf1 = cat8(*(const v8bf*)brow1, *(const v8bf*)(brow1 + 16));

      acc0 = __builtin_amdgcn_wmma_f32_16x16x32_bf16(
          false, af, false, bf0, (short)0, acc0, false, false);
      acc1 = __builtin_amdgcn_wmma_f32_16x16x32_bf16(
          false, af, false, bf1, (short)0, acc1, false, false);
    }
    __syncthreads();
  }

  // epilogue: VGPR i -> M = mb+i, N = l15 (per 16-wide tile)
  int mb = m_block + wm * 16 + half * 8;
  int n0 = n_block + wn * 32 + l15;
  if constexpr (OUT_F32_MN) {
    float* Y = (float*)Yv;
#pragma unroll
    for (int i = 0; i < 8; ++i) {
      int m = mb + i;
      float sc = g[m], bi = bias[m], sh = be[m];
      float y0 = fmaxf(sc * (acc0[i] + bi) + sh, 0.0f);
      float y1 = fmaxf(sc * (acc1[i] + bi) + sh, 0.0f);
      size_t row = ((size_t)b * M + m) * N;
      Y[row + n0]      = y0;
      Y[row + n0 + 16] = y1;
    }
  } else {
    v8bf o0, o1;
#pragma unroll
    for (int i = 0; i < 8; ++i) {
      int m = mb + i;
      float sc = g[m], bi = bias[m], sh = be[m];
      o0[i] = (__bf16)fmaxf(sc * (acc0[i] + bi) + sh, 0.0f);
      o1[i] = (__bf16)fmaxf(sc * (acc1[i] + bi) + sh, 0.0f);
    }
    __bf16* Y = (__bf16*)Yv;
    *(v8bf*)&Y[((size_t)b * N + n0) * M + mb]        = o0;   // packed b128 store
    *(v8bf*)&Y[((size_t)b * N + n0 + 16) * M + mb]   = o1;
  }
}

// ---------------------------------------------------------------------------
// Host-side pipeline
// ---------------------------------------------------------------------------
extern "C" void kernel_launch(void* const* d_in, const int* in_sizes, int n_in,
                              void* d_out, int out_size, void* d_ws, size_t ws_size,
                              hipStream_t stream) {
  (void)in_sizes; (void)n_in; (void)out_size; (void)ws_size;

  const float* p[5] = {(const float*)d_in[0], (const float*)d_in[2],
                       (const float*)d_in[4], (const float*)d_in[6],
                       (const float*)d_in[8]};
  const float* f[5] = {(const float*)d_in[1], (const float*)d_in[3],
                       (const float*)d_in[5], (const float*)d_in[7],
                       (const float*)d_in[9]};
  const int*   cls  = (const int*)d_in[10];
  const float* gc1W = (const float*)d_in[11];
  const float* gc1b = (const float*)d_in[12];
  const float* gc2W = (const float*)d_in[13];
  const float* gc2b = (const float*)d_in[14];
  // stages[s][l] params at d_in[15 + s*20 + l*4 + {W,b,g,be}]

  const int NS[5]  = {8192, 2048, 512, 128, 32};
  const int stK[4]  = {434, 384, 768, 1536};   // layer-0 Cin (actual)
  const int stKp[4] = {448, 384, 768, 1536};   // layer-0 Cin (padded, mult 64)
  const int stM[4]  = {64, 128, 256, 512};     // Cout (all layers)

  // ---- workspace layout ----
  float* fws  = (float*)d_ws;
  size_t off  = 0;
  float* emb1 = fws + off; off += BATCH * 64;
  float* emb2 = fws + off; off += BATCH * 128;
  int*   idxb = (int*)(fws + off); off += (size_t)BATCH * 8192 * 3;
  float* wb   = fws + off; off += (size_t)BATCH * 8192 * 3;
  __bf16* bufX = (__bf16*)(fws + off); off += (size_t)BATCH * 8192 * 448 / 2;
  __bf16* bufP = (__bf16*)(fws + off); off += (size_t)BATCH * 8192 * 64 / 2 + 64;
  __bf16* bufQ = (__bf16*)(fws + off); off += (size_t)BATCH * 8192 * 64 / 2 + 64;
  __bf16* wbuf = (__bf16*)(fws + off);

  // bf16 weight pointers + conversion launches
  __bf16* wp[4][5];
  size_t woff = 0;
  for (int s = 0; s < 4; ++s) {
    for (int l = 0; l < 5; ++l) {
      int K  = l ? stM[s] : stK[s];
      int Kp = l ? stM[s] : stKp[s];
      int M  = stM[s];
      wp[s][l] = wbuf + woff;
      woff += (size_t)M * Kp;
      convert_w_kernel<<<(M * Kp + 255) / 256, 256, 0, stream>>>(
          (const float*)d_in[15 + s * 20 + l * 4], wp[s][l], M, K, Kp);
    }
  }

  auto gemm = [&](const __bf16* Xp, int s, int l, int Kp, int Nn, void* Yp,
                  bool f32out) {
    int M = stM[s];
    int widx = 15 + s * 20 + l * 4;
    dim3 grid(Nn / BN, M / BM, BATCH);
    const float* bb = (const float*)d_in[widx + 1];
    const float* gg = (const float*)d_in[widx + 2];
    const float* ee = (const float*)d_in[widx + 3];
    if (f32out)
      gemm_bn_relu_kernel<true><<<grid, 256, 0, stream>>>(
          Xp, wp[s][l], bb, gg, ee, Yp, M, Kp, Nn);
    else
      gemm_bn_relu_kernel<false><<<grid, 256, 0, stream>>>(
          Xp, wp[s][l], bb, gg, ee, Yp, M, Kp, Nn);
  };
  auto knn = [&](int l1, int l2) {
    knn3_kernel<<<(BATCH * NS[l1] + 255) / 256, 256, 0, stream>>>(
        p[l1], p[l2], NS[l1], NS[l2], idxb, wb);
  };

  embed_kernel<<<BATCH, 256, 0, stream>>>(f[3], f[4], gc1W, gc1b, gc2W, gc2b,
                                          emb1, emb2);

  // ---- stage 3: f3u [4][128][512] (bf16 NC) ----
  knn(3, 4);
  {
    size_t tot = (size_t)BATCH * 128 * 1536;
    build_x_kernel<<<(tot + 255) / 256, 256, 0, stream>>>(
        f[3], 512, f[4], 1024, /*srcNC=*/0, 0, idxb, wb, 128, 32, 1536, bufX);
  }
  gemm(bufX, 3, 0, 1536, 128, bufP, false);
  gemm(bufP, 3, 1, 512, 128, bufQ, false);
  gemm(bufQ, 3, 2, 512, 128, bufP, false);
  gemm(bufP, 3, 3, 512, 128, bufQ, false);
  gemm(bufQ, 3, 4, 512, 128, bufP, false);          // f3u = bufP

  // ---- stage 2: f2u [4][512][256] ----
  knn(2, 3);
  {
    size_t tot = (size_t)BATCH * 512 * 768;
    build_x_kernel<<<(tot + 255) / 256, 256, 0, stream>>>(
        f[2], 256, bufP, 512, /*srcNC=*/1, 512, idxb, wb, 512, 128, 768, bufX);
  }
  gemm(bufX, 2, 0, 768, 512, bufQ, false);
  gemm(bufQ, 2, 1, 256, 512, bufP, false);
  gemm(bufP, 2, 2, 256, 512, bufQ, false);
  gemm(bufQ, 2, 3, 256, 512, bufP, false);
  gemm(bufP, 2, 4, 256, 512, bufQ, false);          // f2u = bufQ

  // ---- stage 1: f1u [4][2048][128] ----
  knn(1, 2);
  {
    size_t tot = (size_t)BATCH * 2048 * 384;
    build_x_kernel<<<(tot + 255) / 256, 256, 0, stream>>>(
        f[1], 128, bufQ, 256, /*srcNC=*/1, 256, idxb, wb, 2048, 512, 384, bufX);
  }
  gemm(bufX, 1, 0, 384, 2048, bufP, false);
  gemm(bufP, 1, 1, 128, 2048, bufQ, false);
  gemm(bufQ, 1, 2, 128, 2048, bufP, false);
  gemm(bufP, 1, 3, 128, 2048, bufQ, false);
  gemm(bufQ, 1, 4, 128, 2048, bufP, false);         // f1u = bufP

  // ---- stage 0: f0u -> d_out f32 [4][64][8192] ----
  knn(0, 1);
  {
    size_t tot = (size_t)BATCH * 8192 * 448;
    build_x0_kernel<<<(tot + 255) / 256, 256, 0, stream>>>(
        emb1, emb2, cls, f[0], bufP, 128, idxb, wb, 8192, 2048, 448, bufX);
  }
  gemm(bufX, 0, 0, 448, 8192, bufQ, false);
  gemm(bufQ, 0, 1, 64, 8192, bufP, false);
  gemm(bufP, 0, 2, 64, 8192, bufQ, false);
  gemm(bufQ, 0, 3, 64, 8192, bufP, false);
  gemm(bufP, 0, 4, 64, 8192, d_out, true);          // final layer -> output
}
